// EncoderLayer_71287867179274
// MI455X (gfx1250) — compile-verified, matrix operand order checked by reading, not run on hardware
//
#include <hip/hip_runtime.h>
#include <hip/hip_bf16.h>
#include <stdint.h>

// ---------------- constants ----------------
#define BATCH   4
#define S_LEN   2048
#define HEADS_N 16
#define DKV     64
#define DMODEL  1024
#define DFF     4096
#define ROWS    (BATCH * S_LEN)   // 8192
#define LN_EPS  1e-5f

typedef unsigned short u16;
typedef __attribute__((ext_vector_type(16))) __bf16 v16bf;
typedef __attribute__((ext_vector_type(8)))  float  v8f;
typedef __attribute__((ext_vector_type(4)))  unsigned int v4u;

union Frag32 { v4u u[2]; v16bf v; };

// round-to-nearest-even f32 -> bf16
__device__ __forceinline__ u16 f2bf(float f) {
  unsigned int u = __float_as_uint(f);
  u += 0x7FFFu + ((u >> 16) & 1u);
  return (u16)(u >> 16);
}

// Load one 16x32 bf16 operand fragment (A-layout, row-major source, stride ld elems).
// lane l: row = l&15 ; chunk0 = k[(l>>4)*8 .. +7] -> vgpr0-3 ; chunk1 = k[16+(l>>4)*8 ..] -> vgpr4-7
__device__ __forceinline__ v16bf load_frag(const u16* __restrict__ p, int ld) {
  const int l = threadIdx.x & 31;
  const u16* row = p + (size_t)(l & 15) * ld + ((l >> 4) * 8);
  Frag32 f;
  f.u[0] = *(const v4u*)(row);
  f.u[1] = *(const v4u*)(row + 16);
  return f.v;
}

__device__ __forceinline__ v8f wmma_bf16(v16bf a, v16bf b, v8f c) {
  return __builtin_amdgcn_wmma_f32_16x16x32_bf16(
      /*neg_a=*/false, a, /*neg_b=*/false, b,
      /*c_mod=*/(short)0, c, /*reuse_a=*/false, /*reuse_b=*/false);
}

// ---- CDNA5 async global->LDS copy (ASYNCcnt path), per-lane 16B ----
__device__ __forceinline__ void async_ld_b128(unsigned int lds_byte, const u16* gsrc) {
  asm volatile("global_load_async_to_lds_b128 %0, %1, off"
               :
               : "v"(lds_byte), "v"(gsrc)
               : "memory");
}
__device__ __forceinline__ void wait_async_0() { asm volatile("s_wait_asynccnt 0" ::: "memory"); }
__device__ __forceinline__ void wait_async_2() { asm volatile("s_wait_asynccnt 2" ::: "memory"); }
__device__ __forceinline__ void wait_async_5() { asm volatile("s_wait_asynccnt 5" ::: "memory"); }

__device__ __forceinline__ unsigned int lds_off(const void* p) {
  return (unsigned int)(uintptr_t)p;   // generic ptr to __shared__: low 32 bits = LDS offset
}

// ---------------- elementwise convert ----------------
__global__ void cvt_bf16_kernel(const float* __restrict__ in, u16* __restrict__ out, int n) {
  int i = blockIdx.x * 256 + threadIdx.x;
  if (i < n) out[i] = f2bf(in[i]);
}

// ---------------- transpose + convert: W[K,N] f32 -> Wt[N,K] bf16 ----------------
__global__ void transpose_bf16_kernel(const float* __restrict__ W, u16* __restrict__ Wt,
                                      int Kdim, int Ndim) {
  __shared__ u16 tile[32][33];
  const int bx = blockIdx.x * 32;       // n block
  const int by = blockIdx.y * 32;       // k block
  const int tx = threadIdx.x & 31;
  const int ty = threadIdx.x >> 5;      // 0..7
#pragma unroll
  for (int i = 0; i < 32; i += 8) {
    tile[ty + i][tx] = f2bf(W[(size_t)(by + ty + i) * Ndim + (bx + tx)]);
  }
  __syncthreads();
#pragma unroll
  for (int i = 0; i < 32; i += 8) {
    Wt[(size_t)(bx + ty + i) * Kdim + (by + tx)] = tile[tx][ty + i];
  }
}

// ---------------- WMMA GEMM: C[M,N] = A[M,K] * Bt[N,K]^T + bias ----------------
// Double-buffered async-to-LDS staging, pointer-bump addressing, peeled pipeline
// (no branches in the steady-state inner loop).
// MODE 0: fp32 out | MODE 1: bf16 out + ReLU | MODE 2: bf16 [B,H,S,dk] | MODE 3: bf16 [B,H,dk,S]
template <int MODE>
__global__ void gemm_bf16_kernel(const u16* __restrict__ A, const u16* __restrict__ Bt,
                                 const float* __restrict__ bias, void* __restrict__ outp,
                                 int M, int N, int K) {
  __shared__ __align__(16) u16 Atile[2][64][40];    // buf stride 5120 B
  __shared__ __align__(16) u16 Btile[2][256][40];   // buf stride 20480 B

  const int w    = threadIdx.x >> 5;
  const int lane = threadIdx.x & 31;
  const int rowblk = blockIdx.y * 64;
  const int colblk = blockIdx.x * 256;
  const int row0 = (w >> 2) * 32;
  const int col0 = (w & 3) * 64;

  // --- per-wave staging state: fixed LDS slots + bumped global pointers ---
  const int cA = w * 32 + lane;                 // A: 256 chunks (64 rows x 4 x 16B)
  const int rA = cA >> 2, sA = (cA & 3) * 8;
  const unsigned ldsA = lds_off(&Atile[0][rA][sA]);
  const u16* gA = A + (size_t)(rowblk + rA) * K + sA;

  unsigned ldsB[4];
  const u16* gB[4];
#pragma unroll
  for (int i = 0; i < 4; ++i) {                 // B: 1024 chunks (256 rows x 4 x 16B)
    const int c = (w * 4 + i) * 32 + lane;
    const int r = c >> 2, s = (c & 3) * 8;
    ldsB[i] = lds_off(&Btile[0][r][s]);
    gB[i] = Bt + (size_t)(colblk + r) * K + s;
  }

  auto stage = [&](int buf) {
    const unsigned ao = buf ? (unsigned)sizeof(Atile[0]) : 0u;
    const unsigned bo = buf ? (unsigned)sizeof(Btile[0]) : 0u;
    async_ld_b128(ldsA + ao, gA);
    gA += 32;
#pragma unroll
    for (int i = 0; i < 4; ++i) {
      async_ld_b128(ldsB[i] + bo, gB[i]);
      gB[i] += 32;
    }
  };

  v8f acc[2][4] = {};
  auto compute = [&](int buf) {
    v16bf a0 = load_frag(&Atile[buf][row0][0], 40);
    v16bf a1 = load_frag(&Atile[buf][row0 + 16][0], 40);
#pragma unroll
    for (int j = 0; j < 4; ++j) {
      v16bf bf = load_frag(&Btile[buf][col0 + j * 16][0], 40);
      acc[0][j] = wmma_bf16(a0, bf, acc[0][j]);
      acc[1][j] = wmma_bf16(a1, bf, acc[1][j]);
    }
  };

  stage(0);                          // k = 0
  int buf = 0;
  for (int k = 0; k < K - 32; k += 32) {   // steady state: always stages k+32
    stage(buf ^ 1);
    wait_async_5();
    __syncthreads();
    compute(buf);
    __syncthreads();
    buf ^= 1;
  }
  wait_async_0();                    // peeled last step
  __syncthreads();
  compute(buf);

  // epilogue: C layout -> lane holds col (lane&15), rows r + 8*(lane>=16)
  const int n_local = lane & 15;
  const int roff    = (lane >> 4) * 8;
  const int grow0   = rowblk + row0;
  const int gcol0   = colblk + col0;
#pragma unroll
  for (int j = 0; j < 4; ++j) {
    const int col = gcol0 + j * 16 + n_local;
    const float bv = bias ? bias[col] : 0.0f;
#pragma unroll
    for (int i = 0; i < 2; ++i) {
#pragma unroll
      for (int r = 0; r < 8; ++r) {
        const int row = grow0 + i * 16 + roff + r;
        float val = acc[i][j][r] + bv;
        if (MODE == 0) {
          ((float*)outp)[(size_t)row * N + col] = val;
        } else if (MODE == 1) {
          val = val > 0.0f ? val : 0.0f;
          ((u16*)outp)[(size_t)row * N + col] = f2bf(val);
        } else if (MODE == 2) {
          const int b = row >> 11, s = row & (S_LEN - 1);
          const int h = col >> 6, d = col & (DKV - 1);
          ((u16*)outp)[((((size_t)b * HEADS_N + h) * S_LEN + s) << 6) + d] = f2bf(val);
        } else { // MODE 3
          const int b = row >> 11, s = row & (S_LEN - 1);
          const int h = col >> 6, d = col & (DKV - 1);
          ((u16*)outp)[(((size_t)b * HEADS_N + h) * DKV + d) * S_LEN + s] = f2bf(val);
        }
      }
    }
  }
}

// ---------------- flash attention ----------------
// Q,K : [B,H,S,dk] bf16 ; Vt : [B,H,dk,S] bf16 ; mask : [B,1,1,S] int
// out : [B,S,D] bf16. 8 waves/block share one (b,h); K/V tiles staged once per
// block via async-to-LDS (double buffered, pointer-bump, peeled pipeline).
__global__ void flash_attn_kernel(const u16* __restrict__ Q, const u16* __restrict__ Kb,
                                  const u16* __restrict__ Vt, const int* __restrict__ mask,
                                  u16* __restrict__ out) {
  __shared__ __align__(16) u16   Ktile[2][32][72];  // buf stride 4608 B
  __shared__ __align__(16) u16   Vtile[2][64][40];  // buf stride 5120 B
  __shared__ __align__(16) float sS[8][16][32];
  __shared__ __align__(16) u16   sP[8][16][32];
  __shared__ float sC[8][16];
  __shared__ int   sM[32];

  const int w    = threadIdx.x >> 5;
  const int lane = threadIdx.x & 31;
  const int gw   = blockIdx.x * 8 + w;
  const int qt   = gw & (S_LEN / 16 - 1);
  const int bh   = gw >> 7;                 // same for all 8 waves in a block
  const int b    = bh >> 4;
  const int h    = bh & (HEADS_N - 1);
  const int q0   = qt * 16;

  const u16* Qbase = Q + ((size_t)bh * S_LEN + q0) * DKV;
  const v16bf qf0 = load_frag(Qbase, DKV);        // dk 0..31
  const v16bf qf1 = load_frag(Qbase + 32, DKV);   // dk 32..63

  // --- staging state ---
  const int cK = w * 32 + lane;             // 256 chunks: 32 rows x 8 x 16B
  const int rK = cK >> 3, sK = (cK & 7) * 8;
  const unsigned ldsK = lds_off(&Ktile[0][rK][sK]);
  const u16* gK = Kb + ((size_t)bh * S_LEN + rK) * DKV + sK;

  const int cV = w * 32 + lane;             // 256 chunks: 64 rows x 4 x 16B
  const int rV = cV >> 2, sV = (cV & 3) * 8;
  const unsigned ldsV = lds_off(&Vtile[0][rV][sV]);
  const u16* gV = Vt + ((size_t)bh * DKV + rV) * S_LEN + sV;

  auto stage = [&](int buf) {
    const unsigned ko = buf ? (unsigned)sizeof(Ktile[0]) : 0u;
    const unsigned vo = buf ? (unsigned)sizeof(Vtile[0]) : 0u;
    async_ld_b128(ldsK + ko, gK);
    gK += 32 * DKV;                          // next 32 key rows
    async_ld_b128(ldsV + vo, gV);
    gV += 32;                                // next 32 key columns
  };

  v8f O[4] = {};
  float m_run = -1e30f, l_run = 0.0f;
  const int* mrow = mask + (size_t)b * S_LEN;

  const int n_local = lane & 15;
  const int roff    = (lane >> 4) * 8;

  auto body = [&](int buf, int kbase) {
    // mask broadcast: one coalesced read by wave 0
    if (w == 0) sM[lane] = mrow[kbase + lane];

    // --- scores: 16(q) x 32(k) from LDS K tile, 1/sqrt(dk) folded in ---
#pragma unroll
    for (int j = 0; j < 2; ++j) {
      v16bf k0 = load_frag(&Ktile[buf][j * 16][0], 72);
      v16bf k1 = load_frag(&Ktile[buf][j * 16][32], 72);
      v8f s = {};
      s = wmma_bf16(qf0, k0, s);
      s = wmma_bf16(qf1, k1, s);
#pragma unroll
      for (int r = 0; r < 8; ++r)
        sS[w][roff + r][j * 16 + n_local] = s[r] * 0.125f;
    }
    __syncthreads();

    // --- online softmax, two passes over LDS (no big register arrays) ---
    if (lane < 16) {
      float mx = m_run;
#pragma unroll
      for (int c = 0; c < 32; ++c) {         // pass 1: mask + max, write back
        float x = sS[w][lane][c];
        if (sM[c] == 0) x = -3.0e38f;
        sS[w][lane][c] = x;
        mx = fmaxf(mx, x);
      }
      const float corr = __expf(m_run - mx);
      float ls = l_run * corr;
#pragma unroll
      for (int c = 0; c < 32; ++c) {         // pass 2: exp + sum + P tile
        const float p = __expf(sS[w][lane][c] - mx);
        ls += p;
        sP[w][lane][c] = f2bf(p);
      }
      m_run = mx;
      l_run = ls;
      sC[w][lane] = corr;
    }
    __syncthreads();

    // --- rescale O and accumulate P x V from LDS V tile ---
    float cr[8];
#pragma unroll
    for (int r = 0; r < 8; ++r) cr[r] = sC[w][roff + r];
    v16bf pf = load_frag(&sP[w][0][0], 32);
#pragma unroll
    for (int d = 0; d < 4; ++d) {
      v16bf vf = load_frag(&Vtile[buf][d * 16][0], 40);
#pragma unroll
      for (int r = 0; r < 8; ++r) O[d][r] = O[d][r] * cr[r];
      O[d] = wmma_bf16(pf, vf, O[d]);
    }
    __syncthreads();   // buf fully consumed before re-staging
  };

  stage(0);                                  // kbase = 0
  int buf = 0;
  for (int kbase = 0; kbase < S_LEN - 32; kbase += 32) {
    stage(buf ^ 1);                          // stages kbase+32, no branch
    wait_async_2();
    __syncthreads();
    body(buf, kbase);
    buf ^= 1;
  }
  wait_async_0();                            // peeled last block
  __syncthreads();
  body(buf, S_LEN - 32);

  // --- final 1/l scaling + store bf16 to [B,S,D] ---
  if (lane < 16) sC[w][lane] = 1.0f / l_run;
  __syncthreads();
  float inv[8];
#pragma unroll
  for (int r = 0; r < 8; ++r) inv[r] = sC[w][roff + r];
#pragma unroll
  for (int d = 0; d < 4; ++d) {
#pragma unroll
    for (int r = 0; r < 8; ++r) {
      const int q = q0 + roff + r;
      out[((size_t)(b * S_LEN + q)) * DMODEL + h * DKV + d * 16 + n_local] =
          f2bf(O[d][r] * inv[r]);
    }
  }
}

// ---------------- fused residual + LayerNorm ----------------
__global__ void add_ln_kernel(const float* __restrict__ x, const float* __restrict__ br,
                              const float* __restrict__ g, const float* __restrict__ be,
                              float* __restrict__ out, u16* __restrict__ out_bf) {
  __shared__ float red[256];
  __shared__ float stat[2];
  const int row = blockIdx.x;
  const int t = threadIdx.x;
  const float* xr = x + (size_t)row * DMODEL;
  const float* yr = br + (size_t)row * DMODEL;

  float v[4];
  float s1 = 0.0f;
#pragma unroll
  for (int i = 0; i < 4; ++i) {
    const int c = t + i * 256;
    v[i] = xr[c] + yr[c];
    s1 += v[i];
  }
  red[t] = s1;
  __syncthreads();
  for (int s = 128; s > 0; s >>= 1) {
    if (t < s) red[t] += red[t + s];
    __syncthreads();
  }
  if (t == 0) stat[0] = red[0] * (1.0f / DMODEL);
  __syncthreads();
  const float mean = stat[0];

  float s2 = 0.0f;
#pragma unroll
  for (int i = 0; i < 4; ++i) {
    const float d = v[i] - mean;
    s2 += d * d;
  }
  red[t] = s2;
  __syncthreads();
  for (int s = 128; s > 0; s >>= 1) {
    if (t < s) red[t] += red[t + s];
    __syncthreads();
  }
  if (t == 0) stat[1] = __frsqrt_rn(red[0] * (1.0f / DMODEL) + LN_EPS);
  __syncthreads();
  const float rstd = stat[1];

#pragma unroll
  for (int i = 0; i < 4; ++i) {
    const int c = t + i * 256;
    const float o = (v[i] - mean) * rstd * g[c] + be[c];
    out[(size_t)row * DMODEL + c] = o;
    if (out_bf) out_bf[(size_t)row * DMODEL + c] = f2bf(o);
  }
}

// ---------------- launch ----------------
extern "C" void kernel_launch(void* const* d_in, const int* in_sizes, int n_in,
                              void* d_out, int out_size, void* d_ws, size_t ws_size,
                              hipStream_t stream) {
  const float* x    = (const float*)d_in[0];
  const int*   mask = (const int*)  d_in[1];
  const float* Wq = (const float*)d_in[2];  const float* bq = (const float*)d_in[3];
  const float* Wk = (const float*)d_in[4];  const float* bk = (const float*)d_in[5];
  const float* Wv = (const float*)d_in[6];  const float* bv = (const float*)d_in[7];
  const float* Wo = (const float*)d_in[8];  const float* bo = (const float*)d_in[9];
  const float* W1 = (const float*)d_in[10]; const float* b1 = (const float*)d_in[11];
  const float* W2 = (const float*)d_in[12]; const float* b2 = (const float*)d_in[13];
  const float* g1 = (const float*)d_in[14]; const float* be1 = (const float*)d_in[15];
  const float* g2 = (const float*)d_in[16]; const float* be2 = (const float*)d_in[17];

  char* ws = (char*)d_ws;
  size_t off = 0;
  u16* XB  = (u16*)(ws + off); off += (size_t)ROWS * DMODEL * 2;       // x bf16
  u16* WQT = (u16*)(ws + off); off += (size_t)DMODEL * DMODEL * 2;
  u16* WKT = (u16*)(ws + off); off += (size_t)DMODEL * DMODEL * 2;
  u16* WVT = (u16*)(ws + off); off += (size_t)DMODEL * DMODEL * 2;
  u16* WOT = (u16*)(ws + off); off += (size_t)DMODEL * DMODEL * 2;
  u16* W1T = (u16*)(ws + off); off += (size_t)DMODEL * DFF * 2;        // [DFF, DMODEL]
  u16* W2T = (u16*)(ws + off); off += (size_t)DFF * DMODEL * 2;        // [DMODEL, DFF]
  u16* QB  = (u16*)(ws + off); off += (size_t)ROWS * DMODEL * 2;       // [B,H,S,dk]
  u16* KB  = (u16*)(ws + off); off += (size_t)ROWS * DMODEL * 2;       // [B,H,S,dk]
  u16* VTB = (u16*)(ws + off); off += (size_t)ROWS * DMODEL * 2;       // [B,H,dk,S]
  u16* AOB = (u16*)(ws + off); off += (size_t)ROWS * DMODEL * 2;       // attn out bf16
  float* YF  = (float*)(ws + off); off += (size_t)ROWS * DMODEL * 4;   // GEMM fp32 out (reused)
  float* X1F = (float*)(ws + off); off += (size_t)ROWS * DMODEL * 4;   // post-LN1 fp32
  u16*   X1B = (u16*)(ws + off);   off += (size_t)ROWS * DMODEL * 2;   // post-LN1 bf16
  u16*   HB  = (u16*)(ws + off);   off += (size_t)ROWS * DFF * 2;      // ReLU hidden bf16

  // 1) convert x, transpose/convert weights
  cvt_bf16_kernel<<<(ROWS * DMODEL) / 256, 256, 0, stream>>>(x, XB, ROWS * DMODEL);
  transpose_bf16_kernel<<<dim3(DMODEL / 32, DMODEL / 32), 256, 0, stream>>>(Wq, WQT, DMODEL, DMODEL);
  transpose_bf16_kernel<<<dim3(DMODEL / 32, DMODEL / 32), 256, 0, stream>>>(Wk, WKT, DMODEL, DMODEL);
  transpose_bf16_kernel<<<dim3(DMODEL / 32, DMODEL / 32), 256, 0, stream>>>(Wv, WVT, DMODEL, DMODEL);
  transpose_bf16_kernel<<<dim3(DMODEL / 32, DMODEL / 32), 256, 0, stream>>>(Wo, WOT, DMODEL, DMODEL);
  transpose_bf16_kernel<<<dim3(DFF / 32, DMODEL / 32), 256, 0, stream>>>(W1, W1T, DMODEL, DFF);
  transpose_bf16_kernel<<<dim3(DMODEL / 32, DFF / 32), 256, 0, stream>>>(W2, W2T, DFF, DMODEL);

  // 2) QKV projections (WMMA) with scatter epilogues
  dim3 gD(DMODEL / 256, ROWS / 64);   // (4, 128)
  gemm_bf16_kernel<2><<<gD, 256, 0, stream>>>(XB, WQT, bq, QB, ROWS, DMODEL, DMODEL);
  gemm_bf16_kernel<2><<<gD, 256, 0, stream>>>(XB, WKT, bk, KB, ROWS, DMODEL, DMODEL);
  gemm_bf16_kernel<3><<<gD, 256, 0, stream>>>(XB, WVT, bv, VTB, ROWS, DMODEL, DMODEL);

  // 3) flash attention (WMMA) -> AOB [B,S,D] bf16
  flash_attn_kernel<<<(BATCH * HEADS_N * (S_LEN / 16)) / 8, 256, 0, stream>>>(QB, KB, VTB, mask, AOB);

  // 4) output projection + residual LN1
  gemm_bf16_kernel<0><<<gD, 256, 0, stream>>>(AOB, WOT, bo, YF, ROWS, DMODEL, DMODEL);
  add_ln_kernel<<<ROWS, 256, 0, stream>>>(x, YF, g1, be1, X1F, X1B);

  // 5) FFN + residual LN2 -> d_out
  dim3 gF(DFF / 256, ROWS / 64);      // (16, 128)
  gemm_bf16_kernel<1><<<gF, 256, 0, stream>>>(X1B, W1T, b1, HB, ROWS, DFF, DMODEL);
  gemm_bf16_kernel<0><<<gD, 256, 0, stream>>>(HB, W2T, b2, YF, ROWS, DMODEL, DFF);
  add_ln_kernel<<<ROWS, 256, 0, stream>>>(X1F, YF, g2, be2, (float*)d_out, nullptr);

  (void)in_sizes; (void)n_in; (void)out_size; (void)ws_size;
}
